// MY_expGCN_parallel_model_19645180412814
// MI455X (gfx1250) — matrix operand order repaired and checked by reading, not compiled, not run on hardware
//
#include <hip/hip_runtime.h>
#include <math.h>

typedef __attribute__((ext_vector_type(16))) _Float16 v16h;
typedef __attribute__((ext_vector_type(8)))  float    v8f;
typedef __attribute__((ext_vector_type(4)))  float    v4f;

#define BATCH   256
#define N_CHEM  12800
#define N_EXP   250368
#define E_CHEM  51200
#define E_EXP   1502208
#define BN_EPS  1e-5f

__device__ __forceinline__ void atomAddF(float* p, float v) {
  __hip_atomic_fetch_add(p, v, __ATOMIC_RELAXED, __HIP_MEMORY_SCOPE_AGENT);
}

__device__ __forceinline__ float elu1(float v) { return v > 0.0f ? v : expm1f(v); }

// ---------------------------------------------------------------------------
// WMMA GEMM: Y[N x O] = X[N x K] @ W[K x O]  (W pre-converted to f16, stored
// transposed as WT[O x K]).  Each wave computes a 16(M) x 64(N) strip:
// 4 v_wmma_f32_16x16x32_f16 per 32-wide K chunk.  Optional bias + ELU.
// Requires N%16==0, K%32==0, O%64==0.
// ---------------------------------------------------------------------------
__global__ void k_wmma_gemm(const float* __restrict__ X, const _Float16* __restrict__ WT,
                            float* __restrict__ Y, int N, int K, int O,
                            const float* __restrict__ bias, int act) {
  const int lane  = threadIdx.x & 31;
  const int wave  = threadIdx.x >> 5;
  const int tilesG = O >> 6;                       // groups of 64 output cols
  const int nTiles = (N >> 4) * tilesG;
  int t = blockIdx.x * (blockDim.x >> 5) + wave;
  if (t >= nTiles) return;                         // wave-uniform: EXEC stays full
  const int tm  = t / tilesG;
  const int tg  = t % tilesG;
  const int l15 = lane & 15;
  const int hi  = lane >> 4;                       // K half-group selector

  const float* xrow = X + (size_t)(tm * 16 + l15) * K + hi * 8;
  v8f acc[4] = {};
  for (int k0 = 0; k0 < K; k0 += 32) {
    // A fragment: lane<16 holds K {k0..k0+7, k0+16..k0+23}; lane>=16 offset +8
    const float* xa = xrow + k0;
    v4f a0 = *(const v4f*)(xa);
    v4f a1 = *(const v4f*)(xa + 4);
    v4f a2 = *(const v4f*)(xa + 16);
    v4f a3 = *(const v4f*)(xa + 20);
    v16h a;
#pragma unroll
    for (int j = 0; j < 4; ++j) {
      a[j]      = (_Float16)a0[j];
      a[j + 4]  = (_Float16)a1[j];
      a[j + 8]  = (_Float16)a2[j];
      a[j + 12] = (_Float16)a3[j];
    }
    // B fragments: lane = column, lanes 0-15 hold K {k0..k0+7, k0+16..k0+23}; lanes 16-31 +8
    const int kb = k0 + hi * 8;
#pragma unroll
    for (int nt = 0; nt < 4; ++nt) {
      const _Float16* wrow = WT + (size_t)(tg * 64 + nt * 16 + l15) * K + kb;
      typedef __attribute__((ext_vector_type(8))) _Float16 v8h_t;
      v8h_t b0 = *(const v8h_t*)wrow;
      v8h_t b1 = *(const v8h_t*)(wrow + 16);
      v16h b = __builtin_shufflevector(b0, b1, 0,1,2,3,4,5,6,7,8,9,10,11,12,13,14,15);
      acc[nt] = __builtin_amdgcn_wmma_f32_16x16x32_f16(
          false, a, false, b, (short)0, acc[nt], false, false);
    }
  }
  // C/D layout: VGPR r -> (M = hi*8 + r, N = lane&15)
  const int m0 = tm * 16 + hi * 8;
#pragma unroll
  for (int nt = 0; nt < 4; ++nt) {
    const int c = tg * 64 + nt * 16 + l15;
    const float bv = bias ? bias[c] : 0.0f;
#pragma unroll
    for (int r = 0; r < 8; ++r) {
      float v = acc[nt][r] + bv;
      if (act == 1) v = elu1(v);
      Y[(size_t)(m0 + r) * O + c] = v;
    }
  }
}

// convert W[K x O] f32 -> WT[O x K] f16
__global__ void k_cvt_wt(const float* __restrict__ W, _Float16* __restrict__ WT, int K, int O) {
  int gid = blockIdx.x * blockDim.x + threadIdx.x;
  if (gid >= K * O) return;
  int k = gid / O, o = gid % O;
  WT[(size_t)o * K + k] = (_Float16)W[gid];
}

// Y[N x 64] = X[N x 3] @ W[3 x 64]   (exp layer 0; K too small for WMMA)
__global__ void k_gemm_k3(const float* __restrict__ X, const float* __restrict__ W,
                          float* __restrict__ Y, int n, int o) {
  int gid = blockIdx.x * blockDim.x + threadIdx.x;
  if (gid >= n * o) return;
  int r = gid / o, c = gid % o;
  const float* xr = X + (size_t)r * 3;
  Y[gid] = xr[0] * W[c] + xr[1] * W[o + c] + xr[2] * W[2 * o + c];
}

// ------------------------- degree / normalization --------------------------
__global__ void k_fill1(float* p, int n) {
  int i = blockIdx.x * blockDim.x + threadIdx.x;
  if (i < n) p[i] = 1.0f;                           // self-loop weight
}
__global__ void k_deg(float* deg, const int* __restrict__ dst,
                      const float* __restrict__ w, int E) {
  int e = blockIdx.x * blockDim.x + threadIdx.x;
  if (e >= E) return;
  atomAddF(&deg[dst[e]], w ? w[e] : 1.0f);
}
__global__ void k_rsqrt(float* d, int n) {
  int i = blockIdx.x * blockDim.x + threadIdx.x;
  if (i >= n) return;
  float v = d[i];
  d[i] = v > 0.0f ? rsqrtf(v) : 0.0f;
}

// ------------------------- GCN aggregation ---------------------------------
// out[n] = dis[n]^2 * xw[n] + bias   (self loop, w=1)
__global__ void k_agg_init(float* __restrict__ out, const float* __restrict__ xw,
                           const float* __restrict__ dis, const float* __restrict__ bias,
                           int n, int o) {
  int gid = blockIdx.x * blockDim.x + threadIdx.x;
  if (gid >= n * o) return;
  int r = gid / o, c = gid % o;
  float s = dis[r];
  out[gid] = s * s * xw[gid] + bias[c];
}
// out[d] += dis[s]*w*dis[d] * xw[s]  — 4 feats/thread, f32 global atomics,
// prefetch a future edge's source row (random-gather lookahead into L2/L0).
__global__ void k_agg_edges(float* __restrict__ out, const float* __restrict__ xw,
                            const float* __restrict__ dis, const int* __restrict__ src,
                            const int* __restrict__ dst, const float* __restrict__ w,
                            int E, int o) {
  const int per = o >> 2;
  int gid = blockIdx.x * blockDim.x + threadIdx.x;
  if (gid >= E * per) return;
  int e = gid / per;
  int f = (gid - e * per) << 2;
  int s = src[e], d = dst[e];
  int e2 = e + 4096;
  if (e2 < E) __builtin_prefetch(xw + (size_t)src[e2] * o + f, 0, 3);
  float coef = dis[s] * (w ? w[e] : 1.0f) * dis[d];
  v4f v = *(const v4f*)(xw + (size_t)s * o + f);
  float* op = out + (size_t)d * o + f;
  atomAddF(op + 0, coef * v[0]);
  atomAddF(op + 1, coef * v[1]);
  atomAddF(op + 2, coef * v[2]);
  atomAddF(op + 3, coef * v[3]);
}

// ------------------------- BatchNorm (training stats) ----------------------
__global__ void k_zero(float* p, int n) {
  int i = blockIdx.x * blockDim.x + threadIdx.x;
  if (i < n) p[i] = 0.0f;
}
__global__ void k_bn_stats(const float* __restrict__ x, float* __restrict__ stats,
                           int n, int o) {        // o divides 256
  __shared__ float ss[256], sq[256];
  const int tid = threadIdx.x;
  const int col = tid % o;
  const int rpb = 256 / o;
  float s = 0.0f, q = 0.0f;
  for (int r = blockIdx.x * rpb + tid / o; r < n; r += gridDim.x * rpb) {
    float v = x[(size_t)r * o + col];
    s += v; q += v * v;
  }
  ss[tid] = s; sq[tid] = q;
  __syncthreads();
  if (tid < o) {
    float S = ss[tid], Q = sq[tid];
    for (int k = 1; k < rpb; ++k) { S += ss[tid + k * o]; Q += sq[tid + k * o]; }
    atomAddF(&stats[tid], S);
    atomAddF(&stats[o + tid], Q);
  }
}
__global__ void k_bn_elu(float* __restrict__ x, const float* __restrict__ stats,
                         const float* __restrict__ g, const float* __restrict__ be,
                         int n, int o) {
  int gid = blockIdx.x * blockDim.x + threadIdx.x;
  if (gid >= n * o) return;
  int c = gid % o;
  float inv_n = 1.0f / (float)n;
  float m = stats[c] * inv_n;
  float var = stats[o + c] * inv_n - m * m;
  float y = (x[gid] - m) * rsqrtf(var + BN_EPS) * g[c] + be[c];
  x[gid] = elu1(y);
}

// ------------------------- pool / concat / final ---------------------------
__global__ void k_pool_tanh(const float* __restrict__ x, float* __restrict__ y,
                            int b, int p, int f) {
  int gid = blockIdx.x * blockDim.x + threadIdx.x;
  if (gid >= b * f) return;
  int g = gid / f, c = gid % f;
  const float* base = x + (size_t)g * p * f + c;
  float s = 0.0f;
  for (int i = 0; i < p; ++i) s += base[(size_t)i * f];
  y[gid] = tanhf(s / (float)p);
}
__global__ void k_concat2(const float* __restrict__ a, const float* __restrict__ b,
                          float* __restrict__ y, int n, int fa, int fb) {
  int gid = blockIdx.x * blockDim.x + threadIdx.x;
  int ft = fa + fb;
  if (gid >= n * ft) return;
  int r = gid / ft, c = gid % ft;
  y[gid] = (c < fa) ? a[(size_t)r * fa + c] : b[(size_t)r * fb + (c - fa)];
}
__global__ void k_final_dot(const float* __restrict__ X, const float* __restrict__ w,
                            const float* __restrict__ b, float* __restrict__ out,
                            int n, int k) {
  int i = blockIdx.x * blockDim.x + threadIdx.x;
  if (i >= n) return;
  float s = b[0];
  for (int j = 0; j < k; ++j) s += X[(size_t)i * k + j] * w[j];
  out[i] = s;
}

// ---------------------------------------------------------------------------
static inline int cdiv(long long a, long long b) { return (int)((a + b - 1) / b); }

static void gcn_layer(const float* xin, float* xw, float* out, int N, int K, int O,
                      const float* W, const float* bias, const int* src, const int* dst,
                      const float* ew, int E, const float* dis, _Float16* wh,
                      hipStream_t s) {
  k_cvt_wt<<<cdiv((long long)K * O, 256), 256, 0, s>>>(W, wh, K, O);
  int tiles = (N / 16) * (O / 64);
  k_wmma_gemm<<<cdiv(tiles, 8), 256, 0, s>>>(xin, wh, xw, N, K, O, nullptr, 0);
  k_agg_init<<<cdiv((long long)N * O, 256), 256, 0, s>>>(out, xw, dis, bias, N, O);
  k_agg_edges<<<cdiv((long long)E * (O / 4), 256), 256, 0, s>>>(out, xw, dis, src, dst, ew, E, O);
}

static void bn_elu(float* x, int N, int O, const float* g, const float* be,
                   float* stats, hipStream_t s) {
  k_zero<<<1, 256, 0, s>>>(stats, 2 * O);
  k_bn_stats<<<512, 256, 0, s>>>(x, stats, N, O);
  k_bn_elu<<<cdiv((long long)N * O, 256), 256, 0, s>>>(x, stats, g, be, N, O);
}

static void fc_head(const float* cp, const float* ep, float* cat, float* t0, float* dout,
                    const float* W0, const float* b0, const float* W1, const float* b1,
                    _Float16* wh, hipStream_t s) {
  k_concat2<<<cdiv(BATCH * 128, 256), 256, 0, s>>>(cp, ep, cat, BATCH, 64, 64);
  k_cvt_wt<<<cdiv(128 * 128, 256), 256, 0, s>>>(W0, wh, 128, 128);
  k_wmma_gemm<<<cdiv((BATCH / 16) * 2, 8), 256, 0, s>>>(cat, wh, t0, BATCH, 128, 128, b0, 1);
  k_cvt_wt<<<cdiv(128 * 128, 256), 256, 0, s>>>(W1, wh, 128, 128);
  k_wmma_gemm<<<cdiv((BATCH / 16) * 2, 8), 256, 0, s>>>(t0, wh, dout, BATCH, 128, 128, b1, 0);
}

extern "C" void kernel_launch(void* const* d_in, const int* in_sizes, int n_in,
                              void* d_out, int out_size, void* d_ws, size_t ws_size,
                              hipStream_t stream) {
  (void)in_sizes; (void)n_in; (void)out_size; (void)ws_size;
  const float* Drug1_F = (const float*)d_in[0];
  const float* Drug2_F = (const float*)d_in[1];
  const int*   ADJ1    = (const int*)d_in[2];
  const int*   ADJ2    = (const int*)d_in[3];
  const float* EXP1    = (const float*)d_in[4];
  const float* EXP2    = (const float*)d_in[5];
  const int*   EADJ    = (const int*)d_in[6];
  const float* EWGT    = (const float*)d_in[7];
  const float* cW0 = (const float*)d_in[9],  *cW1 = (const float*)d_in[10], *cW2 = (const float*)d_in[11];
  const float* cb0 = (const float*)d_in[12], *cb1 = (const float*)d_in[13], *cb2 = (const float*)d_in[14];
  const float* cg0 = (const float*)d_in[15], *cg1 = (const float*)d_in[16];
  const float* cB0 = (const float*)d_in[17], *cB1 = (const float*)d_in[18];
  const float* eW0 = (const float*)d_in[19], *eW1 = (const float*)d_in[20], *eW2 = (const float*)d_in[21];
  const float* eb0 = (const float*)d_in[22], *eb1 = (const float*)d_in[23], *eb2 = (const float*)d_in[24];
  const float* eg0 = (const float*)d_in[25], *eg1 = (const float*)d_in[26];
  const float* eB0 = (const float*)d_in[27], *eB1 = (const float*)d_in[28];
  const float* f1W0 = (const float*)d_in[29], *f1W1 = (const float*)d_in[30];
  const float* f1b0 = (const float*)d_in[31], *f1b1 = (const float*)d_in[32];
  const float* sW0 = (const float*)d_in[33], *sW1 = (const float*)d_in[34], *sW2 = (const float*)d_in[35];
  const float* sb0 = (const float*)d_in[36], *sb1 = (const float*)d_in[37], *sb2 = (const float*)d_in[38];

  // workspace layout (floats)
  float* ws = (float*)d_ws;
  const size_t NE64 = (size_t)N_EXP * 64;
  float* big0  = ws;                       // node features (h)
  float* big1  = big0 + NE64;              // xw scratch
  float* dis_e = big1 + NE64;
  float* dis_1 = dis_e + N_EXP;
  float* dis_2 = dis_1 + N_CHEM;
  float* stats = dis_2 + N_CHEM;           // 256
  float* c1p = stats + 256;                // BATCH x 64 each
  float* c2p = c1p + BATCH * 64;
  float* e1p = c2p + BATCH * 64;
  float* e2p = e1p + BATCH * 64;
  float* d1  = e2p + BATCH * 64;           // BATCH x 128
  float* d2  = d1 + BATCH * 128;
  float* cat = d2 + BATCH * 128;           // BATCH x 256
  float* t0  = cat + BATCH * 256;
  float* t1  = t0 + BATCH * 256;
  _Float16* wh = (_Float16*)(t1 + BATCH * 256);   // up to 256x256 halves

  // ---- symmetric-norm degree factors ----
  k_fill1<<<cdiv(N_EXP, 256), 256, 0, stream>>>(dis_e, N_EXP);
  k_deg<<<cdiv(E_EXP, 256), 256, 0, stream>>>(dis_e, EADJ + E_EXP, EWGT, E_EXP);
  k_rsqrt<<<cdiv(N_EXP, 256), 256, 0, stream>>>(dis_e, N_EXP);
  k_fill1<<<cdiv(N_CHEM, 256), 256, 0, stream>>>(dis_1, N_CHEM);
  k_deg<<<cdiv(E_CHEM, 256), 256, 0, stream>>>(dis_1, ADJ1 + E_CHEM, nullptr, E_CHEM);
  k_rsqrt<<<cdiv(N_CHEM, 256), 256, 0, stream>>>(dis_1, N_CHEM);
  k_fill1<<<cdiv(N_CHEM, 256), 256, 0, stream>>>(dis_2, N_CHEM);
  k_deg<<<cdiv(E_CHEM, 256), 256, 0, stream>>>(dis_2, ADJ2 + E_CHEM, nullptr, E_CHEM);
  k_rsqrt<<<cdiv(N_CHEM, 256), 256, 0, stream>>>(dis_2, N_CHEM);

  // ---- chem branches (shared weights) ----
  const float* dF[2]  = {Drug1_F, Drug2_F};
  const int*   dA[2]  = {ADJ1, ADJ2};
  const float* dD[2]  = {dis_1, dis_2};
  float*       cP[2]  = {c1p, c2p};
  for (int i = 0; i < 2; ++i) {
    const int* src = dA[i]; const int* dst = dA[i] + E_CHEM;
    gcn_layer(dF[i], big1, big0, N_CHEM, 64, 128, cW0, cb0, src, dst, nullptr, E_CHEM, dD[i], wh, stream);
    bn_elu(big0, N_CHEM, 128, cg0, cB0, stats, stream);
    gcn_layer(big0, big1, big0, N_CHEM, 128, 128, cW1, cb1, src, dst, nullptr, E_CHEM, dD[i], wh, stream);
    bn_elu(big0, N_CHEM, 128, cg1, cB1, stats, stream);
    gcn_layer(big0, big1, big0, N_CHEM, 128, 64, cW2, cb2, src, dst, nullptr, E_CHEM, dD[i], wh, stream);
    k_pool_tanh<<<cdiv(BATCH * 64, 256), 256, 0, stream>>>(big0, cP[i], BATCH, 50, 64);
  }

  // ---- expression branches (shared weights + adjacency) ----
  const float* xE[2] = {EXP1, EXP2};
  float*       eP[2] = {e1p, e2p};
  const int* esrc = EADJ; const int* edst = EADJ + E_EXP;
  for (int i = 0; i < 2; ++i) {
    // layer 0: K=3 scalar GEMM
    k_gemm_k3<<<cdiv((long long)N_EXP * 64, 256), 256, 0, stream>>>(xE[i], eW0, big1, N_EXP, 64);
    k_agg_init<<<cdiv((long long)N_EXP * 64, 256), 256, 0, stream>>>(big0, big1, dis_e, eb0, N_EXP, 64);
    k_agg_edges<<<cdiv((long long)E_EXP * 16, 256), 256, 0, stream>>>(big0, big1, dis_e, esrc, edst, EWGT, E_EXP, 64);
    bn_elu(big0, N_EXP, 64, eg0, eB0, stats, stream);
    gcn_layer(big0, big1, big0, N_EXP, 64, 64, eW1, eb1, esrc, edst, EWGT, E_EXP, dis_e, wh, stream);
    bn_elu(big0, N_EXP, 64, eg1, eB1, stats, stream);
    gcn_layer(big0, big1, big0, N_EXP, 64, 64, eW2, eb2, esrc, edst, EWGT, E_EXP, dis_e, wh, stream);
    k_pool_tanh<<<cdiv(BATCH * 64, 256), 256, 0, stream>>>(big0, eP[i], BATCH, 978, 64);
  }

  // ---- heads ----
  fc_head(c1p, e1p, cat, t0, d1, f1W0, f1b0, f1W1, f1b1, wh, stream);
  fc_head(c2p, e2p, cat, t0, d2, f1W0, f1b0, f1W1, f1b1, wh, stream);

  k_concat2<<<cdiv(BATCH * 256, 256), 256, 0, stream>>>(d1, d2, cat, BATCH, 128, 128);
  k_cvt_wt<<<cdiv(256 * 256, 256), 256, 0, stream>>>(sW0, wh, 256, 256);
  k_wmma_gemm<<<cdiv((BATCH / 16) * 4, 8), 256, 0, stream>>>(cat, wh, t0, BATCH, 256, 256, sb0, 1);
  k_cvt_wt<<<cdiv(256 * 128, 256), 256, 0, stream>>>(sW1, wh, 256, 128);
  k_wmma_gemm<<<cdiv((BATCH / 16) * 2, 8), 256, 0, stream>>>(t0, wh, t1, BATCH, 256, 128, sb1, 1);
  k_final_dot<<<1, 256, 0, stream>>>(t1, sW2, sb2, (float*)d_out, BATCH, 128);
}
